// UNetVTEncoder_71614284693899
// MI455X (gfx1250) — matrix operand order repaired and checked by reading, not compiled, not run on hardware
//
#include <hip/hip_runtime.h>
#include <hip/hip_bf16.h>

// ---------------------------------------------------------------------------
// CDNA5 (gfx1250) Swin-style windowed-attention block.
// All heavy math via v_wmma_f32_16x16x32_bf16 (f32 accumulate).
// Every A/B fragment operand (including the 3x3-conv im2col, via a tap-major
// K ordering + channel-last window image) is a contiguous 32B run per lane
// (global_load_b128 / ds_load_b128) in k = 16*half + s order, same in A and B.
// ---------------------------------------------------------------------------

typedef __bf16 bf16_t;
typedef __attribute__((ext_vector_type(16))) __bf16 bf16x16;
typedef __attribute__((ext_vector_type(8)))  __bf16 bf16x8;
typedef __attribute__((ext_vector_type(8)))  float  f32x8;

#define LANE() ((int)(threadIdx.x & 31))

__device__ inline f32x8 wmma_bf16(bf16x16 a, bf16x16 b, f32x8 c) {
  return __builtin_amdgcn_wmma_f32_16x16x32_bf16(
      false, a, false, b, (short)0, c, false, false);
}

__device__ inline bf16x16 cat16(bf16x8 lo, bf16x8 hi) {
  return __builtin_shufflevector(lo, hi, 0, 1, 2, 3, 4, 5, 6, 7,
                                 8, 9, 10, 11, 12, 13, 14, 15);
}

// A fragment: 16x32 (MxK) bf16, row-major, leading dim ld (elements).
// Layout: fragment elements 0..15 are K = 16*half..16*half+15 of row m
// (matches B fragments) — one contiguous 32B run per lane.
__device__ inline bf16x16 frag_a_ld(const bf16_t* p, int ld) {
  int lane = LANE();
  int m = lane & 15, half = lane >> 4;
  const bf16_t* r0 = p + (size_t)m * ld + 16 * half;
  bf16x8 lo = *(const bf16x8*)(r0);
  bf16x8 hi = *(const bf16x8*)(r0 + 8);
  return cat16(lo, hi);
}

// B fragment: 32x16 (KxN) where memory is N-major: element (k,n) at
// p[n*ld + k]. Per lane: one contiguous 16-element run at K = 16*half.
__device__ inline bf16x16 frag_b_tr(const bf16_t* p, int ld) {
  int lane = LANE();
  int n = lane & 15, half = lane >> 4;
  const bf16_t* r0 = p + (size_t)n * ld + 16 * half;
  bf16x8 lo = *(const bf16x8*)(r0);
  bf16x8 hi = *(const bf16x8*)(r0 + 8);
  return cat16(lo, hi);
}

// Clamped scalar-gather variant (attention V tail only: 1 of 13 iterations).
// Clamping keeps reads in-bounds so garbage/NaN can't poison the WMMA.
__device__ inline bf16x16 frag_b_tr_clamp(const bf16_t* p, int ld, int kmax) {
  int lane = LANE();
  int n = lane & 15, half = lane >> 4;
  bf16x16 b;
#pragma unroll
  for (int i = 0; i < 16; ++i) {
    int k = i + 16 * half;
    if (k > kmax) k = kmax;
    b[i] = p[(size_t)n * ld + k];
  }
  return b;
}

// Row reductions for the C/D layout: a row lives in one vgpr across one
// 16-lane half; xor masks 1/2/4/8 never cross the half boundary on wave32.
__device__ inline float rowmax16(float v) {
  v = fmaxf(v, __shfl_xor(v, 1));
  v = fmaxf(v, __shfl_xor(v, 2));
  v = fmaxf(v, __shfl_xor(v, 4));
  v = fmaxf(v, __shfl_xor(v, 8));
  return v;
}
__device__ inline float rowsum16(float v) {
  v += __shfl_xor(v, 1);
  v += __shfl_xor(v, 2);
  v += __shfl_xor(v, 4);
  v += __shfl_xor(v, 8);
  return v;
}

// ---------------------------------------------------------------------------
// Prep kernels
// ---------------------------------------------------------------------------
__global__ void f32_to_bf16_kernel(const float* __restrict__ in,
                                   bf16_t* __restrict__ out, int n) {
  int i = blockIdx.x * 256 + threadIdx.x;
  if (i < n) out[i] = (bf16_t)in[i];
}

// qkv weight permutation to tap-major K: W[o][(kh*3+kw)*256 + ic]
__global__ void qkvw_permute_kernel(const float* __restrict__ in,
                                    bf16_t* __restrict__ out) {
  int i = blockIdx.x * 256 + threadIdx.x;
  if (i >= 768 * 2304) return;
  int o = i / 2304, rem = i % 2304;
  int tap = rem >> 8, ic = rem & 255;
  out[i] = (bf16_t)in[((size_t)o * 256 + ic) * 9 + tap];
}

__global__ __launch_bounds__(512) void kpm_kernel(
    const unsigned char* __restrict__ pm, unsigned char* __restrict__ kpm) {
  __shared__ int cnt;
  int win = blockIdx.x;
  int b = win / 144, wr = (win / 12) % 12, wc = win % 12;
  if (threadIdx.x == 0) cnt = 0;
  __syncthreads();
  int p = threadIdx.x;
  unsigned char m = 0;
  if (p < 400) {
    int y = wr * 20 + p / 20, x = wc * 20 + p % 20;
    m = pm[(size_t)b * 57600 + y * 240 + x] ? 1 : 0;
    if (m) atomicAdd(&cnt, 1);
  }
  __syncthreads();
  if (p < 400) kpm[win * 400 + p] = (cnt == 400) ? 0 : m;
}

// LayerNorm over C (per pixel) -> zero-padded channel-last window image
// xw[win][22][22][256] (interior at +1,+1). One block per (b,row).
// Per-pixel writes are 512 contiguous bytes.
__global__ __launch_bounds__(256) void ln1_window_kernel(
    const float* __restrict__ src, const float* __restrict__ g,
    const float* __restrict__ be, bf16_t* __restrict__ xw) {
  int row = blockIdx.x;
  int b = row / 240, y = row % 240;
  int x = threadIdx.x;
  if (x >= 240) return;
  const float* base = src + ((size_t)b * 256 * 240 + y) * 240 + x;
  float s = 0.f, s2 = 0.f;
  for (int c = 0; c < 256; ++c) {
    float v = base[(size_t)c * 57600];
    s += v; s2 += v * v;
  }
  float mean = s * (1.0f / 256.0f);
  float var = s2 * (1.0f / 256.0f) - mean * mean;
  float rs = rsqrtf(var + 1e-5f);
  int win = b * 144 + (y / 20) * 12 + x / 20;
  int wy = y % 20, wx = x % 20;
  bf16_t* dst = xw + ((size_t)win * 484 + (wy + 1) * 22 + (wx + 1)) * 256;
  for (int c = 0; c < 256; ++c) {
    float v = base[(size_t)c * 57600];
    dst[c] = (bf16_t)((v - mean) * rs * g[c] + be[c]);
  }
}

// LayerNorm2 from y1 (NCHW f32) -> token-major bf16 x2[pix][256]
__global__ __launch_bounds__(256) void ln2_kernel(
    const float* __restrict__ y1, const float* __restrict__ g,
    const float* __restrict__ be, bf16_t* __restrict__ x2) {
  int row = blockIdx.x;
  int b = row / 240, y = row % 240;
  int x = threadIdx.x;
  if (x >= 240) return;
  const float* base = y1 + ((size_t)b * 256 * 240 + y) * 240 + x;
  float s = 0.f, s2 = 0.f;
  for (int c = 0; c < 256; ++c) {
    float v = base[(size_t)c * 57600];
    s += v; s2 += v * v;
  }
  float mean = s * (1.0f / 256.0f);
  float var = s2 * (1.0f / 256.0f) - mean * mean;
  float rs = rsqrtf(var + 1e-5f);
  size_t pix = (size_t)b * 57600 + y * 240 + x;
  bf16_t* dst = x2 + pix * 256;
  for (int c = 0; c < 256; ++c) {
    float v = base[(size_t)c * 57600];
    dst[c] = (bf16_t)((v - mean) * rs * g[c] + be[c]);
  }
}

// ---------------------------------------------------------------------------
// QKV 3x3 conv as implicit GEMM per window: O[768][400] = W[768][2304] * col
// K ordering is tap-major (k = tap*256 + ic) and the window image is
// channel-last, so each lane's 16-element K run is contiguous (2x b128).
// ---------------------------------------------------------------------------
__device__ inline bf16x16 frag_b_im2col(const bf16_t* xwin, int k0, int n0) {
  int lane = LANE();
  int n = lane & 15, half = lane >> 4;
  int p = n0 + n, py = p / 20, px = p % 20;
  int kk = k0 + 16 * half;       // 16-run start; never crosses a tap boundary
  int tap = kk >> 8, ic0 = kk & 255;
  int kh = tap / 3, kw = tap - kh * 3;
  const bf16_t* r0 = xwin + (size_t)((py + kh) * 22 + px + kw) * 256 + ic0;
  bf16x8 lo = *(const bf16x8*)(r0);
  bf16x8 hi = *(const bf16x8*)(r0 + 8);
  return cat16(lo, hi);
}

__global__ __launch_bounds__(256) void qkv_kernel(
    const bf16_t* __restrict__ Wq, const bf16_t* __restrict__ xw,
    const float* __restrict__ qkvb, bf16_t* __restrict__ q,
    bf16_t* __restrict__ kk, bf16_t* __restrict__ vt) {
  int wave = threadIdx.x >> 5, lane = LANE();
  int win = blockIdx.x / 48;
  int m0 = (blockIdx.x % 48) * 16;
  int part = m0 >> 8;              // uniform per block: 0=q, 1=k, 2=v
  int chb = m0 & 255;              // uniform channel base
  const bf16_t* xwin = xw + (size_t)win * 484 * 256;
  int n = lane & 15, half = lane >> 4;
  for (int nt = wave; nt < 25; nt += 8) {
    int n0 = nt * 16;
    f32x8 acc = {};
    for (int k0 = 0; k0 < 2304; k0 += 32) {
      bf16x16 a = frag_a_ld(Wq + (size_t)m0 * 2304 + k0, 2304);
      bf16x16 b = frag_b_im2col(xwin, k0, n0);
      acc = wmma_bf16(a, b, acc);
    }
    int p = n0 + n;
    if (part == 0) {                       // Q: [win,h][tok][d], pre-scaled
#pragma unroll
      for (int r = 0; r < 8; ++r) {
        int ch = chb + r + 8 * half;
        int h = ch >> 5, d = ch & 31;
        float val = (acc[r] + qkvb[m0 + r + 8 * half]) * 0.17677669529663687f;
        q[(((size_t)win * 8 + h) * 400 + p) * 32 + d] = (bf16_t)val;
      }
    } else if (part == 1) {                // K: [win,h][tok][d]
#pragma unroll
      for (int r = 0; r < 8; ++r) {
        int ch = chb + r + 8 * half;
        int h = ch >> 5, d = ch & 31;
        float val = acc[r] + qkvb[m0 + r + 8 * half];
        kk[(((size_t)win * 8 + h) * 400 + p) * 32 + d] = (bf16_t)val;
      }
    } else {                               // V transposed: [win,h][d][tok]
#pragma unroll
      for (int r = 0; r < 8; ++r) {
        int ch = chb + r + 8 * half;
        int h = ch >> 5, d = ch & 31;
        float val = acc[r] + qkvb[m0 + r + 8 * half];
        vt[(((size_t)win * 8 + h) * 32 + d) * 400 + p] = (bf16_t)val;
      }
    }
  }
}

// ---------------------------------------------------------------------------
// Flash-style windowed attention: one wave per 16-row Q tile of one
// (window, head); streams K/V tiles in pairs (K-dim 32 per PV WMMA) with
// online softmax. P re-layouted via conflict-free padded LDS scratch.
// ---------------------------------------------------------------------------
#define PROWS 40   // padded row stride of P scratch (80B: bank-conflict-free)

__global__ __launch_bounds__(256) void attn_kernel(
    const bf16_t* __restrict__ qg, const bf16_t* __restrict__ kg,
    const bf16_t* __restrict__ vg, const unsigned char* __restrict__ kpm,
    const float* __restrict__ rpb, bf16_t* __restrict__ attnout) {
  __shared__ __align__(16) bf16_t pbuf[8][16 * PROWS];
  int wave = threadIdx.x >> 5, lane = LANE();
  int idx = blockIdx.x * 8 + wave;       // 0 .. 57599
  int qt = idx % 25;
  int h = (idx / 25) & 7;
  int win = idx / 200;
  const bf16_t* qb = qg + (size_t)(win * 8 + h) * 400 * 32;
  const bf16_t* kb = kg + (size_t)(win * 8 + h) * 400 * 32;
  const bf16_t* vt = vg + (size_t)(win * 8 + h) * 32 * 400;
  const unsigned char* kpw = kpm + win * 400;
  int n = lane & 15, half = lane >> 4;

  bf16x16 aq = frag_a_ld(qb + qt * 16 * 32, 32);   // Q tile [16 tok][32]

  // loop-invariant query-token coordinates per accumulator register
  int whi[8], wwi[8];
#pragma unroll
  for (int r = 0; r < 8; ++r) {
    int i = qt * 16 + r + 8 * half;
    whi[r] = i / 20;
    wwi[r] = i % 20;
  }

  f32x8 o0 = {}, o1 = {};
  float mrun[8], lrun[8];
#pragma unroll
  for (int r = 0; r < 8; ++r) { mrun[r] = -1e30f; lrun[r] = 0.f; }
  bf16_t* pb = pbuf[wave];

  for (int kt = 0; kt < 25; kt += 2) {
    bool tail = (kt == 24);
    f32x8 s0 = {}, s1 = {};
    { bf16x16 bk = frag_b_tr(kb + kt * 16 * 32, 32);       // K^T tile
      s0 = wmma_bf16(aq, bk, s0); }
    if (!tail) {
      bf16x16 bk = frag_b_tr(kb + (kt + 1) * 16 * 32, 32);
      s1 = wmma_bf16(aq, bk, s1);
    }
    // relative-position bias + key padding mask
    int j0 = kt * 16 + n;
    int wj0 = j0 / 20, xj0 = j0 % 20;
    int j1 = j0 + 16;
    int wj1 = j1 / 20, xj1 = j1 % 20;
    bool km0 = kpw[j0] != 0;
    bool km1 = tail ? true : (kpw[j1] != 0);
#pragma unroll
    for (int r = 0; r < 8; ++r) {
      {
        int dx = wj0 - whi[r]; dx += (dx < 0) ? 39 : 0;
        int dy = xj0 - wwi[r]; dy += (dy < 0) ? 39 : 0;
        s0[r] = km0 ? -1e30f : (s0[r] + rpb[(dx * 39 + dy) * 8 + h]);
      }
      if (!tail) {
        int dx = wj1 - whi[r]; dx += (dx < 0) ? 39 : 0;
        int dy = xj1 - wwi[r]; dy += (dy < 0) ? 39 : 0;
        s1[r] = km1 ? -1e30f : (s1[r] + rpb[(dx * 39 + dy) * 8 + h]);
      } else {
        s1[r] = -1e30f;
      }
    }
    // online softmax (row stats are register-aligned with the C/D layout)
#pragma unroll
    for (int r = 0; r < 8; ++r) {
      float rm = fmaxf(rowmax16(s0[r]), rowmax16(s1[r]));
      float mn = fmaxf(mrun[r], rm);
      float fac = __expf(mrun[r] - mn);
      mrun[r] = mn;
      float p0 = __expf(s0[r] - mn);
      float p1 = __expf(s1[r] - mn);
      lrun[r] = lrun[r] * fac + rowsum16(p0) + rowsum16(p1);
      o0[r] *= fac; o1[r] *= fac;
      int row = r + 8 * half;
      pb[row * PROWS + n]      = (bf16_t)p0;
      pb[row * PROWS + 16 + n] = (bf16_t)p1;
    }
    asm volatile("s_wait_dscnt 0" ::: "memory");   // LDS RAW within wave
    bf16x16 ap = frag_a_ld(pb, PROWS);             // P tile [16][32]
    bf16x16 bv0, bv1;
    if (!tail) {
      bv0 = frag_b_tr(vt + kt * 16, 400);            // d = 0..15
      bv1 = frag_b_tr(vt + 16 * 400 + kt * 16, 400); // d = 16..31
    } else {   // 16 valid tokens; clamp reads, zero P kills the rest
      bv0 = frag_b_tr_clamp(vt + kt * 16, 400, 15);
      bv1 = frag_b_tr_clamp(vt + 16 * 400 + kt * 16, 400, 15);
    }
    o0 = wmma_bf16(ap, bv0, o0);
    o1 = wmma_bf16(ap, bv1, o1);
  }
  // finalize: divide by row sums, write token-major [win][tok][C]
#pragma unroll
  for (int r = 0; r < 8; ++r) {
    float inv = 1.0f / lrun[r];
    int i = qt * 16 + r + 8 * half;
    size_t base = ((size_t)win * 400 + i) * 256 + h * 32;
    attnout[base + n]      = (bf16_t)(o0[r] * inv);
    attnout[base + 16 + n] = (bf16_t)(o1[r] * inv);
  }
}

// ---------------------------------------------------------------------------
// 1x1 out-projection GEMM + window merge + residual: y1 = src + Wout*attn + b
// ---------------------------------------------------------------------------
__global__ __launch_bounds__(256) void outproj_kernel(
    const bf16_t* __restrict__ Wo, const bf16_t* __restrict__ ao,
    const float* __restrict__ ob, const float* __restrict__ src,
    float* __restrict__ y1) {
  int wave = threadIdx.x >> 5, lane = LANE();
  int win = blockIdx.x >> 4;
  int m0 = (blockIdx.x & 15) * 16;
  int b = win / 144, wr = (win / 12) % 12, wc = win % 12;
  int n = lane & 15, half = lane >> 4;
  for (int nt = wave; nt < 25; nt += 8) {
    int n0 = nt * 16;
    f32x8 acc = {};
    for (int k0 = 0; k0 < 256; k0 += 32) {
      bf16x16 a = frag_a_ld(Wo + (size_t)m0 * 256 + k0, 256);
      bf16x16 bb = frag_b_tr(ao + ((size_t)win * 400 + n0) * 256 + k0, 256);
      acc = wmma_bf16(a, bb, acc);
    }
    int p = n0 + n;
    int y = wr * 20 + p / 20, x = wc * 20 + p % 20;
#pragma unroll
    for (int r = 0; r < 8; ++r) {
      int ch = m0 + r + 8 * half;
      size_t addr = (((size_t)b * 256 + ch) * 240 + y) * 240 + x;
      y1[addr] = acc[r] + ob[ch] + src[addr];
    }
  }
}

// ---------------------------------------------------------------------------
// Fused FFN: per 64-pixel tile, h = relu(W1*x + b1) lives in dynamic LDS as
// [64 pixels][2048+8] bf16 (263KB, within CDNA5's 320KB/WGP; padded stride
// -> conflict-free ds_load_b128 B fragments). Then out = W2*h + b2 +
// residual with padding mask. 16 waves per block.
// ---------------------------------------------------------------------------
#define HSTRIDE 2056

__global__ __launch_bounds__(512) void ffn_kernel(
    const bf16_t* __restrict__ W1, const float* __restrict__ b1,
    const bf16_t* __restrict__ W2, const float* __restrict__ b2,
    const bf16_t* __restrict__ x2, const float* __restrict__ y1,
    const unsigned char* __restrict__ pmask, float* __restrict__ out) {
  extern __shared__ __align__(16) char smem_raw[];
  bf16_t* hsm = (bf16_t*)smem_raw;                 // [64][HSTRIDE]
  int wave = threadIdx.x >> 5, lane = LANE();
  int n = lane & 15, half = lane >> 4;
  size_t pix0 = (size_t)blockIdx.x * 64;

  // Phase 1: h = relu(W1 * x + b1), stored pixel-major in LDS
  for (int tile = wave; tile < 512; tile += 16) {
    int m0 = (tile >> 2) * 16;
    int n0 = (tile & 3) * 16;
    f32x8 acc = {};
    for (int k0 = 0; k0 < 256; k0 += 32) {
      bf16x16 a = frag_a_ld(W1 + (size_t)m0 * 256 + k0, 256);
      bf16x16 bb = frag_b_tr(x2 + (pix0 + n0) * 256 + k0, 256);
      acc = wmma_bf16(a, bb, acc);
    }
#pragma unroll
    for (int r = 0; r < 8; ++r) {
      int m = m0 + r + 8 * half;
      float vv = acc[r] + b1[m];
      hsm[(size_t)(n0 + n) * HSTRIDE + m] = (bf16_t)(vv > 0.f ? vv : 0.f);
    }
  }
  __syncthreads();

  // Phase 2: out = W2 * h + b2 + residual, masked
  for (int tile = wave; tile < 64; tile += 16) {
    int m0 = (tile >> 2) * 16;
    int n0 = (tile & 3) * 16;
    f32x8 acc = {};
    for (int k0 = 0; k0 < 2048; k0 += 32) {
      __builtin_prefetch(W2 + (size_t)m0 * 2048 + k0 + 64, 0, 1);
      bf16x16 a = frag_a_ld(W2 + (size_t)m0 * 2048 + k0, 2048);
      bf16x16 bb = frag_b_tr(hsm + (size_t)n0 * HSTRIDE + k0, HSTRIDE);
      acc = wmma_bf16(a, bb, acc);
    }
#pragma unroll
    for (int r = 0; r < 8; ++r) {
      int ch = m0 + r + 8 * half;
      size_t p = pix0 + n0 + n;
      int b = (int)(p / 57600);
      int rem = (int)(p % 57600);
      int y = rem / 240, x = rem % 240;
      size_t addr = (((size_t)b * 256 + ch) * 240 + y) * 240 + x;
      float vv = acc[r] + b2[ch] + y1[addr];
      out[addr] = pmask[p] ? 0.f : vv;
    }
  }
}

// ---------------------------------------------------------------------------
extern "C" void kernel_launch(void* const* d_in, const int* in_sizes, int n_in,
                              void* d_out, int out_size, void* d_ws,
                              size_t ws_size, hipStream_t stream) {
  const float* src  = (const float*)d_in[0];
  const unsigned char* pmask = (const unsigned char*)d_in[1]; // bool mask
  const float* n1w  = (const float*)d_in[2];
  const float* n1b  = (const float*)d_in[3];
  const float* qkvw = (const float*)d_in[4];
  const float* qkvb = (const float*)d_in[5];
  const float* outw = (const float*)d_in[6];
  const float* outb = (const float*)d_in[7];
  const float* rpb  = (const float*)d_in[8];
  const float* n2w  = (const float*)d_in[9];
  const float* n2b  = (const float*)d_in[10];
  const float* l1w  = (const float*)d_in[11];
  const float* l1b  = (const float*)d_in[12];
  const float* l2w  = (const float*)d_in[13];
  const float* l2b  = (const float*)d_in[14];
  float* out = (float*)d_out;

  char* base = (char*)d_ws;
  size_t off = 0;
  auto alloc = [&](size_t bytes) -> void* {
    void* p = base + off;
    off = (off + bytes + 255) & ~((size_t)255);
    return p;
  };
  bf16_t* xw   = (bf16_t*)alloc((size_t)288 * 484 * 256 * 2); // channel-last
  bf16_t* qb   = (bf16_t*)alloc((size_t)288 * 8 * 400 * 32 * 2);
  bf16_t* kbuf = (bf16_t*)alloc((size_t)288 * 8 * 400 * 32 * 2);
  bf16_t* vtb  = (bf16_t*)alloc((size_t)288 * 8 * 32 * 400 * 2); // V^T
  bf16_t* ao   = (bf16_t*)alloc((size_t)288 * 400 * 256 * 2);
  float*  y1   = (float*) alloc((size_t)2 * 256 * 240 * 240 * 4);
  bf16_t* x2   = (bf16_t*)alloc((size_t)115200 * 256 * 2);
  bf16_t* Wqkv = (bf16_t*)alloc((size_t)768 * 2304 * 2);
  bf16_t* Wout = (bf16_t*)alloc((size_t)256 * 256 * 2);
  bf16_t* W1   = (bf16_t*)alloc((size_t)2048 * 256 * 2);
  bf16_t* W2   = (bf16_t*)alloc((size_t)256 * 2048 * 2);
  unsigned char* kpmb = (unsigned char*)alloc((size_t)288 * 400);

  qkvw_permute_kernel<<<(768 * 2304 + 255) / 256, 256, 0, stream>>>(qkvw, Wqkv);
  f32_to_bf16_kernel<<<(256 * 256 + 255) / 256, 256, 0, stream>>>(outw, Wout, 256 * 256);
  f32_to_bf16_kernel<<<(2048 * 256 + 255) / 256, 256, 0, stream>>>(l1w, W1, 2048 * 256);
  f32_to_bf16_kernel<<<(2048 * 256 + 255) / 256, 256, 0, stream>>>(l2w, W2, 2048 * 256);

  // Zero the padded window image (conv padding ring).
  hipMemsetAsync(xw, 0, (size_t)288 * 484 * 256 * 2, stream);

  kpm_kernel<<<288, 512, 0, stream>>>(pmask, kpmb);
  ln1_window_kernel<<<480, 256, 0, stream>>>(src, n1w, n1b, xw);
  qkv_kernel<<<288 * 48, 256, 0, stream>>>(Wqkv, xw, qkvb, qb, kbuf, vtb);
  attn_kernel<<<7200, 256, 0, stream>>>(qb, kbuf, vtb, kpmb, rpb, ao);
  outproj_kernel<<<288 * 16, 256, 0, stream>>>(Wout, ao, outb, src, y1);
  ln2_kernel<<<480, 256, 0, stream>>>(y1, n2w, n2b, x2);

  (void)hipFuncSetAttribute((const void*)ffn_kernel,
                            hipFuncAttributeMaxDynamicSharedMemorySize,
                            64 * HSTRIDE * 2);
  ffn_kernel<<<1800, 512, 64 * HSTRIDE * 2, stream>>>(W1, l1b, W2, l2b, x2, y1,
                                                      pmask, out);
}
